// BiLSTM_Attention_NoIntra_26439818674683
// MI455X (gfx1250) — compile-verified, matrix operand order checked by reading, not run on hardware
//
#include <hip/hip_runtime.h>
#include <hip/hip_bf16.h>

// ---------------------------------------------------------------------------
// Problem constants (from the reference)
// ---------------------------------------------------------------------------
static constexpr int cD  = 32;            // days
static constexpr int cT  = 64;            // topics (seq len of topic LSTM)
static constexpr int cS  = 64;            // sentences (seq len of text LSTM)
static constexpr int cE  = 768;           // embedding
static constexpr int cH  = 128;           // hidden (text/topic LSTMs)
static constexpr int cG  = 4 * cH;        // 512 gates
static constexpr int cHD = 64;            // hidden (day LSTMs)
static constexpr int cGD = 4 * cHD;       // 256 gates
static constexpr int cBT = cD * cT;       // 2048 text-batch
static constexpr int cM1 = cBT * cS;      // 131072 rows of the big GEMM
static constexpr int cBD = 16;            // padded batch for day LSTM (1 -> 16)

typedef __attribute__((ext_vector_type(16))) __bf16 v16bf;
typedef __attribute__((ext_vector_type(8)))  float  v8f;

__device__ __forceinline__ float sigf(float x) { return 1.0f / (1.0f + __expf(-x)); }

// ---------------------------------------------------------------------------
// Pack fp32 row-major [M,K] into bf16 WMMA operand blocks.
// Block (mb,kb) holds a 16x32 tile: 32 lanes x 16 bf16, lane L<16 -> row L,
// K in {0..7,16..23}; lane L>=16 -> row L-16, K in {8..15,24..31}.
// The same layout serves as B-operand blocks when src = W [N,K] row-major
// (since B col-major == Wᵀ col-major == W row-major).
// ---------------------------------------------------------------------------
__global__ void pack_wmma_kernel(const float* __restrict__ src,
                                 __bf16* __restrict__ dst, int M, int K) {
  int tid = blockIdx.x * blockDim.x + threadIdx.x;
  int KB = K >> 5;
  int total = (M >> 4) * KB * 32;          // one thread per (block, lane)
  if (tid >= total) return;
  int lane = tid & 31;
  int blk  = tid >> 5;                     // blk = mb*KB + kb
  int kb   = blk % KB;
  int mb   = blk / KB;
  int row  = mb * 16 + (lane & 15);
  int base = (lane >> 4) * 8;
  v16bf v;
#pragma unroll
  for (int e = 0; e < 16; ++e) {
    int k = kb * 32 + ((e < 8) ? (e + base) : (e + 8 + base));
    v[e] = (__bf16)src[(size_t)row * K + k];
  }
  *(v16bf*)(dst + (size_t)blk * 512 + lane * 16) = v;
}

// ---------------------------------------------------------------------------
// out[M,N] = Apacked[M,K] * Bpacked(K,N) + bias[N]   (bf16 WMMA, fp32 accum)
// One wave per 16x16 output tile.
// ---------------------------------------------------------------------------
__global__ void gemm_bias_kernel(const __bf16* __restrict__ Ap,
                                 const __bf16* __restrict__ Bp,
                                 const float* __restrict__ bias,
                                 float* __restrict__ out, int M, int N, int K) {
  int wid  = (blockIdx.x * blockDim.x + threadIdx.x) >> 5;
  int lane = threadIdx.x & 31;
  int ntiles = N >> 4;
  int mb = wid / ntiles;
  int nb = wid % ntiles;
  if (mb >= (M >> 4)) return;
  int KB = K >> 5;
  const __bf16* pa = Ap + (size_t)mb * KB * 512;
  const __bf16* pb = Bp + (size_t)nb * KB * 512;
  v8f acc = {};
  for (int kb = 0; kb < KB; ++kb) {
    v16bf a = *(const v16bf*)(pa + (size_t)kb * 512 + lane * 16);
    v16bf b = *(const v16bf*)(pb + (size_t)kb * 512 + lane * 16);
    __builtin_prefetch(pa + (size_t)(kb + 1) * 512 + lane * 16, 0, 1);
    acc = __builtin_amdgcn_wmma_f32_16x16x32_bf16(false, a, false, b,
                                                  (short)0, acc, false, false);
  }
  int hi = lane >> 4, l15 = lane & 15;
  int col = nb * 16 + l15;
  float bv = bias ? bias[col] : 0.0f;
#pragma unroll
  for (int r = 0; r < 8; ++r) {
    int row = mb * 16 + r + hi * 8;
    out[(size_t)row * N + col] = acc[r] + bv;
  }
}

// ---------------------------------------------------------------------------
// Recurrent LSTM, Whh resident in LDS (packed bf16 B-layout), h kept in LDS
// in bf16 A-layout.  Xg = precomputed x@Wihᵀ + b, layout [B, S, 4H].
// One wave per 16-row batch tile; `wpb` waves per block share Whh.
// ---------------------------------------------------------------------------
template <int Hc>
__global__ void lstm_rec_kernel(const float* __restrict__ Xg,
                                const __bf16* __restrict__ WhhPacked,
                                float* __restrict__ outSeq,   // [B,S,Hc] or null
                                float* __restrict__ hFinal,   // [B,Hc]   or null
                                int S) {
  constexpr int G  = 4 * Hc;
  constexpr int NB = G / 16;       // gate n-tiles
  constexpr int KB = Hc / 32;      // k-blocks
  constexpr int WELEMS = NB * KB * 512;

  extern __shared__ char smem[];
  __bf16* ldsW = (__bf16*)smem;
  const int wave = threadIdx.x >> 5;
  const int lane = threadIdx.x & 31;
  const int wpb  = blockDim.x >> 5;
  float*  gatesAll = (float*)(ldsW + WELEMS);
  float*  cAll     = gatesAll + (size_t)wpb * 16 * G;
  __bf16* hAAll    = (__bf16*)(cAll + (size_t)wpb * 16 * Hc);
  float*  gates = gatesAll + (size_t)wave * 16 * G;
  float*  cbuf  = cAll + (size_t)wave * 16 * Hc;
  __bf16* hA    = hAAll + (size_t)wave * KB * 512;

  // cooperative 128-bit copy of packed Whh into LDS
  {
    const uint4* sw = (const uint4*)WhhPacked;
    uint4* dw = (uint4*)ldsW;
    for (int i = threadIdx.x; i < (WELEMS * 2) / 16; i += blockDim.x) dw[i] = sw[i];
  }
  for (int i = lane; i < 16 * Hc; i += 32) cbuf[i] = 0.0f;
  for (int i = lane; i < KB * 512; i += 32) hA[i] = (__bf16)0.0f;
  __syncthreads();

  const int mtile = blockIdx.x * wpb + wave;
  const int m0 = mtile * 16;
  const int hi = lane >> 4;
  const int l15 = lane & 15;

  for (int s = 0; s < S; ++s) {
    // gates[16,G] = Xg[:,s,:] + h @ Whhᵀ   (WMMA over LDS operands)
    for (int nb = 0; nb < NB; ++nb) {
      v8f acc;
#pragma unroll
      for (int r = 0; r < 8; ++r) {
        int row = r + hi * 8;
        acc[r] = Xg[(size_t)(m0 + row) * S * G + (size_t)s * G + nb * 16 + l15];
      }
#pragma unroll
      for (int kb = 0; kb < KB; ++kb) {
        v16bf a = *(const v16bf*)(hA + kb * 512 + lane * 16);
        v16bf b = *(const v16bf*)(ldsW + (size_t)(nb * KB + kb) * 512 + lane * 16);
        acc = __builtin_amdgcn_wmma_f32_16x16x32_bf16(false, a, false, b,
                                                      (short)0, acc, false, false);
      }
#pragma unroll
      for (int r = 0; r < 8; ++r) {
        int row = r + hi * 8;
        gates[row * G + nb * 16 + l15] = acc[r];
      }
    }
    __syncthreads();
    // fused elementwise: i,f,g,o -> c,h ; h written back in bf16 A-layout
    for (int t = 0; t < (16 * Hc) / 32; ++t) {
      int flat = t * 32 + lane;
      int row = flat / Hc, col = flat % Hc;
      float gi = gates[row * G + col];
      float gf = gates[row * G + Hc + col];
      float gg = gates[row * G + 2 * Hc + col];
      float go = gates[row * G + 3 * Hc + col];
      float c = sigf(gf) * cbuf[flat] + sigf(gi) * tanhf(gg);
      float h = sigf(go) * tanhf(c);
      cbuf[flat] = c;
      int kb = col >> 5, kk = col & 31;
      int hi2 = (kk >> 3) & 1;
      int e = (kk & 7) + ((kk >> 4) << 3);
      hA[kb * 512 + (row + hi2 * 16) * 16 + e] = (__bf16)h;
      if (outSeq)
        outSeq[(size_t)(m0 + row) * S * Hc + (size_t)s * Hc + col] = h;
      if (hFinal && s == S - 1)
        hFinal[(size_t)(m0 + row) * Hc + col] = h;
    }
    __syncthreads();
  }
}

// ---------------------------------------------------------------------------
// Per-day topic attention with threshold top-k (cumsum-before <= 0.8).
// One block per day (d in [0,32)), 128 threads.
// ---------------------------------------------------------------------------
__global__ void topic_attn_kernel(const float* __restrict__ outP,   // [32,64,128]
                                  const float* __restrict__ hP,     // [32,128]
                                  const float* __restrict__ w1w,    // [128,128]
                                  const float* __restrict__ w1b,    // [128]
                                  float* __restrict__ ctx) {        // [32,128]
  int d = blockIdx.x;
  const float* out = outP + (size_t)d * cT * cH;
  const float* fs = hP + (size_t)d * cH;
  __shared__ float v[cH], score[cT], aw[cT], keepw[cT], sb;
  int tid = threadIdx.x;
  if (tid < cH) {
    float a = 0.0f;
    for (int j = 0; j < cH; ++j) a += fs[j] * w1w[j * cH + tid];
    v[tid] = a;
  }
  if (tid == 0) {
    float a = 0.0f;
    for (int j = 0; j < cH; ++j) a += w1b[j] * fs[j];
    sb = a;
  }
  __syncthreads();
  if (tid < cT) {
    float a = sb;
    for (int k = 0; k < cH; ++k) a += out[tid * cH + k] * v[k];
    score[tid] = a;
  }
  __syncthreads();
  if (tid == 0) {
    float mx = -1e30f;
    for (int t = 0; t < cT; ++t) mx = fmaxf(mx, score[t]);
    float sum = 0.0f;
    for (int t = 0; t < cT; ++t) { aw[t] = __expf(score[t] - mx); sum += aw[t]; }
    for (int t = 0; t < cT; ++t) aw[t] /= sum;
    int used[cT];
    for (int t = 0; t < cT; ++t) used[t] = 0;
    float cum = 0.0f;
    for (int t = 0; t < cT; ++t) {                 // selection sort, descending
      int best = -1; float bv = -1.0f;
      for (int u = 0; u < cT; ++u)
        if (!used[u] && aw[u] > bv) { bv = aw[u]; best = u; }
      used[best] = 1;
      keepw[best] = (cum <= 0.8f) ? bv : 0.0f;
      cum += bv;
    }
  }
  __syncthreads();
  if (tid < cH) {
    float a = 0.0f;
    for (int t = 0; t < cT; ++t) a += keepw[t] * out[t * cH + tid];
    ctx[(size_t)d * cH + tid] = a;
  }
}

// ---------------------------------------------------------------------------
// Day attention + linear heads. Single block of 64 threads.
// ---------------------------------------------------------------------------
__global__ void final_head_kernel(const float* __restrict__ outD,  // [32,64] (b=0 slab)
                                  const float* __restrict__ hD,    // [64]    (b=0 row)
                                  const float* __restrict__ w2w, const float* __restrict__ w2b,
                                  const float* __restrict__ linw, const float* __restrict__ linb,
                                  const float* __restrict__ lin2w, const float* __restrict__ lin2b,
                                  const float* __restrict__ Wp, const float* __restrict__ bp,
                                  const float* __restrict__ Wc, const float* __restrict__ bc,
                                  const float* __restrict__ Wr, const float* __restrict__ br,
                                  const float* __restrict__ Wcpi, const float* __restrict__ bcpi,
                                  const float* __restrict__ plabels,
                                  float* __restrict__ out4) {
  __shared__ float v2[cHD], score[cD], aw[cD], ctx[cHD], z1[48], z[16];
  int tid = threadIdx.x;
  if (tid < cHD) {
    float a = 0.0f;
    for (int j = 0; j < cHD; ++j) a += hD[j] * w2w[j * cHD + tid];
    v2[tid] = a;
  }
  __syncthreads();
  if (tid < cD) {
    float sb = 0.0f;
    for (int j = 0; j < cHD; ++j) sb += w2b[j] * hD[j];
    float a = sb;
    for (int k = 0; k < cHD; ++k) a += outD[tid * cHD + k] * v2[k];
    score[tid] = a;
  }
  __syncthreads();
  if (tid == 0) {
    float mx = -1e30f;
    for (int t = 0; t < cD; ++t) mx = fmaxf(mx, score[t]);
    float sum = 0.0f;
    for (int t = 0; t < cD; ++t) { aw[t] = __expf(score[t] - mx); sum += aw[t]; }
    for (int t = 0; t < cD; ++t) aw[t] /= sum;
  }
  __syncthreads();
  if (tid < cHD) {
    float a = 0.0f;
    for (int t = 0; t < cD; ++t) a += aw[t] * outD[t * cHD + tid];
    ctx[tid] = a;
  }
  __syncthreads();
  if (tid < 48) {
    float a = linb[tid];
    for (int k = 0; k < cHD; ++k) a += ctx[k] * linw[tid * cHD + k];
    z1[tid] = a;
  }
  __syncthreads();
  if (tid < 16) {
    float a = lin2b[tid];
    for (int k = 0; k < 48; ++k) a += z1[k] * lin2w[tid * 48 + k];
    z[tid] = a;
  }
  __syncthreads();
  if (tid < 4) {
    const float* W[4] = {Wp, Wc, Wr, Wcpi};
    const float* B[4] = {bp, bc, br, bcpi};
    float a = B[tid][0];
    for (int k = 0; k < 16; ++k) a += z[k] * W[tid][k];
    for (int k = 0; k < 4; ++k) a += plabels[tid * 4 + k] * W[tid][16 + k];
    out4[tid] = a;
  }
}

__global__ void fill_zero_kernel(float* __restrict__ p, size_t n) {
  size_t i = (size_t)blockIdx.x * blockDim.x + threadIdx.x;
  if (i < n) p[i] = 0.0f;
}

// ---------------------------------------------------------------------------
// Host orchestration
// ---------------------------------------------------------------------------
static inline void launch_pack(const float* src, __bf16* dst, int M, int K,
                               hipStream_t s) {
  int total = (M >> 4) * (K >> 5) * 32;
  pack_wmma_kernel<<<(total + 255) / 256, 256, 0, s>>>(src, dst, M, K);
}
static inline void launch_gemm(const __bf16* A, const __bf16* B, const float* bias,
                               float* out, int M, int N, int K, hipStream_t s) {
  int waves = (M >> 4) * (N >> 4);
  int threads = waves * 32;
  gemm_bias_kernel<<<(threads + 255) / 256, 256, 0, s>>>(A, B, bias, out, M, N, K);
}

extern "C" void kernel_launch(void* const* d_in, const int* in_sizes, int n_in,
                              void* d_out, int out_size, void* d_ws, size_t ws_size,
                              hipStream_t stream) {
  const float* X       = (const float*)d_in[0];
  const float* plabels = (const float*)d_in[1];
  const float* Wih_t0 = (const float*)d_in[4];
  const float* Whh_t0 = (const float*)d_in[5];
  const float* b_t0   = (const float*)d_in[6];
  const float* Wih_p0 = (const float*)d_in[7];
  const float* Whh_p0 = (const float*)d_in[8];
  const float* b_p0   = (const float*)d_in[9];
  const float* Wih_p1 = (const float*)d_in[10];
  const float* Whh_p1 = (const float*)d_in[11];
  const float* b_p1   = (const float*)d_in[12];
  const float* Wih_d0 = (const float*)d_in[13];
  const float* Whh_d0 = (const float*)d_in[14];
  const float* b_d0   = (const float*)d_in[15];
  const float* Wih_d1 = (const float*)d_in[16];
  const float* Whh_d1 = (const float*)d_in[17];
  const float* b_d1   = (const float*)d_in[18];
  const float* w1w = (const float*)d_in[19];
  const float* w1b = (const float*)d_in[20];
  const float* w2w = (const float*)d_in[21];
  const float* w2b = (const float*)d_in[22];
  const float* linw  = (const float*)d_in[23];
  const float* linb  = (const float*)d_in[24];
  const float* lin2w = (const float*)d_in[25];
  const float* lin2b = (const float*)d_in[26];

  char* ws = (char*)d_ws;
  size_t off = 0;
  auto alloc = [&](size_t bytes) -> void* {
    void* p = ws + off;
    off += (bytes + 255) & ~(size_t)255;
    return p;
  };
  // --- workspace layout ---
  __bf16* ApX    = (__bf16*)alloc((size_t)cM1 * cE * 2);      // packed X
  __bf16* BWihT  = (__bf16*)alloc((size_t)cG * cE * 2);       // packed Wih_t0
  __bf16* BWhhT  = (__bf16*)alloc((size_t)cG * cH * 2);       // packed Whh_t0
  float*  XgT    = (float*)alloc((size_t)cM1 * cG * 4);       // gate preacts (text)
  float*  hT     = (float*)alloc((size_t)cBT * cH * 4);       // text final hidden
  __bf16* ApP0   = (__bf16*)alloc((size_t)cBT * cH * 2);
  __bf16* BWihP0 = (__bf16*)alloc((size_t)cG * cH * 2);
  __bf16* BWhhP0 = (__bf16*)alloc((size_t)cG * cH * 2);
  float*  XgP0   = (float*)alloc((size_t)cBT * cG * 4);
  float*  outP0  = (float*)alloc((size_t)cBT * cH * 4);
  float*  hP0    = (float*)alloc((size_t)cD * cH * 4);
  __bf16* ApP1   = (__bf16*)alloc((size_t)cBT * cH * 2);
  __bf16* BWihP1 = (__bf16*)alloc((size_t)cG * cH * 2);
  __bf16* BWhhP1 = (__bf16*)alloc((size_t)cG * cH * 2);
  float*  XgP1   = (float*)alloc((size_t)cBT * cG * 4);
  float*  outP1  = (float*)alloc((size_t)cBT * cH * 4);
  float*  ctx    = (float*)alloc((size_t)cD * cH * 4);
  __bf16* ApCtx  = (__bf16*)alloc((size_t)cD * cH * 2);
  __bf16* BWihD0 = (__bf16*)alloc((size_t)cGD * cH * 2);
  __bf16* BWhhD0 = (__bf16*)alloc((size_t)cGD * cHD * 2);
  float*  XgD0   = (float*)alloc((size_t)cBD * cD * cGD * 4); // padded [16,32,256]
  float*  outD0  = (float*)alloc((size_t)cBD * cD * cHD * 4);
  float*  hD0    = (float*)alloc((size_t)cBD * cHD * 4);
  __bf16* ApD1   = (__bf16*)alloc((size_t)cBD * cD * cHD * 2);
  __bf16* BWihD1 = (__bf16*)alloc((size_t)cGD * cHD * 2);
  __bf16* BWhhD1 = (__bf16*)alloc((size_t)cGD * cHD * 2);
  float*  XgD1   = (float*)alloc((size_t)cBD * cD * cGD * 4);
  float*  outD1  = (float*)alloc((size_t)cBD * cD * cHD * 4);
  (void)ws_size; (void)n_in; (void)in_sizes; (void)out_size;

  // dynamic-LDS sizes for the recurrence kernels
  auto ldsBytes = [](int Hc, int wpb) -> size_t {
    int G = 4 * Hc, KB = Hc / 32, NB = G / 16;
    return (size_t)NB * KB * 512 * 2 +                         // Whh packed
           (size_t)wpb * (16 * G * 4 + 16 * Hc * 4 + KB * 512 * 2);
  };

  // --- Phase 1: text LSTM ---
  launch_pack(X, ApX, cM1, cE, stream);
  launch_pack(Wih_t0, BWihT, cG, cE, stream);
  launch_pack(Whh_t0, BWhhT, cG, cH, stream);
  launch_gemm(ApX, BWihT, b_t0, XgT, cM1, cG, cE, stream);
  lstm_rec_kernel<cH><<<cBT / (16 * 4), 4 * 32, ldsBytes(cH, 4), stream>>>(
      XgT, BWhhT, nullptr, hT, cS);

  // --- Phase 2: topic LSTM p0 ---
  launch_pack(hT, ApP0, cBT, cH, stream);
  launch_pack(Wih_p0, BWihP0, cG, cH, stream);
  launch_pack(Whh_p0, BWhhP0, cG, cH, stream);
  launch_gemm(ApP0, BWihP0, b_p0, XgP0, cBT, cG, cH, stream);
  lstm_rec_kernel<cH><<<1, 2 * 32, ldsBytes(cH, 2), stream>>>(
      XgP0, BWhhP0, outP0, hP0, cT);

  // --- Phase 3: topic LSTM p1 ---
  launch_pack(outP0, ApP1, cBT, cH, stream);
  launch_pack(Wih_p1, BWihP1, cG, cH, stream);
  launch_pack(Whh_p1, BWhhP1, cG, cH, stream);
  launch_gemm(ApP1, BWihP1, b_p1, XgP1, cBT, cG, cH, stream);
  lstm_rec_kernel<cH><<<1, 2 * 32, ldsBytes(cH, 2), stream>>>(
      XgP1, BWhhP1, outP1, nullptr, cT);

  // --- Phase 4: per-day topic attention ---
  topic_attn_kernel<<<cD, 128, 0, stream>>>(outP1, hP0, w1w, w1b, ctx);

  // --- Phase 5: day LSTM d0 (batch 1 padded to 16) ---
  launch_pack(ctx, ApCtx, cD, cH, stream);
  launch_pack(Wih_d0, BWihD0, cGD, cH, stream);
  launch_pack(Whh_d0, BWhhD0, cGD, cHD, stream);
  {
    size_t n = (size_t)cBD * cD * cGD;
    fill_zero_kernel<<<(int)((n + 255) / 256), 256, 0, stream>>>(XgD0, n);
  }
  launch_gemm(ApCtx, BWihD0, b_d0, XgD0, cD, cGD, cH, stream); // writes rows 0..31 (=batch 0)
  lstm_rec_kernel<cHD><<<1, 32, ldsBytes(cHD, 1), stream>>>(
      XgD0, BWhhD0, outD0, hD0, cD);

  // --- Phase 6: day LSTM d1 ---
  launch_pack(outD0, ApD1, cBD * cD, cHD, stream);
  launch_pack(Wih_d1, BWihD1, cGD, cHD, stream);
  launch_pack(Whh_d1, BWhhD1, cGD, cHD, stream);
  launch_gemm(ApD1, BWihD1, b_d1, XgD1, cBD * cD, cGD, cHD, stream);
  lstm_rec_kernel<cHD><<<1, 32, ldsBytes(cHD, 1), stream>>>(
      XgD1, BWhhD1, outD1, nullptr, cD);

  // --- Phase 7: day attention + heads ---
  final_head_kernel<<<1, 64, 0, stream>>>(
      outD1 /* b=0 slab is the first 32*64 floats */, hD0 /* b=0 row */,
      w2w, w2b, linw, linb, lin2w, lin2b,
      (const float*)d_in[27], (const float*)d_in[28],
      (const float*)d_in[29], (const float*)d_in[30],
      (const float*)d_in[31], (const float*)d_in[32],
      (const float*)d_in[33], (const float*)d_in[34],
      plabels, (float*)d_out);
}
